// Generator_7129645712020
// MI455X (gfx1250) — matrix operand order repaired and checked by reading, not compiled, not run on hardware
//
#include <hip/hip_runtime.h>

typedef _Float16 v16h __attribute__((ext_vector_type(16)));
typedef _Float16 v8h  __attribute__((ext_vector_type(8)));
typedef float    v8f  __attribute__((ext_vector_type(8)));

union Frag16 { v16h v; v8h h[2]; };

// Block-cooperative GEMM: sOut[16 x N] = sA[16 x K] @ W[K x N] (+ bias[n]),
// with W prepacked transposed+f16 in global as Wt[n*K + k].
// A: f16 LDS, row-major, row length K (K % 32 == 0, N % 16 == 0).
// 8 waves split the N/16 tiles; one 16x16 f32 accumulator per tile.
template <int K, int N>
__device__ __forceinline__ void gemm16(const _Float16* __restrict__ sA,
                                       const _Float16* __restrict__ Wt,
                                       float* __restrict__ sOut,
                                       const float* __restrict__ bias,
                                       int tid) {
    const int wave = tid >> 5;
    const int lane = tid & 31;
    const int l16  = lane & 15;
    const int hi   = lane >> 4;                 // 0 or 1 (half-wave)
    const _Float16* arow = sA + l16 * K;        // A: M = lane%16
    for (int nt = wave; nt < (N >> 4); nt += 8) {
        const int n = (nt << 4) + l16;          // B/C/D: N = lane%16 within tile
        const _Float16* wrow = Wt + (size_t)n * K;
        v8f acc = {};
#pragma unroll
        for (int k0 = 0; k0 < K; k0 += 32) {
            Frag16 a, b;
            // A 16x32 f16 layout: halves 0-7 -> K = k0+8*hi+0..7,
            //                     halves 8-15 -> K = k0+8*hi+16..23
            a.h[0] = *(const v8h*)(arow + k0 + 8 * hi);
            a.h[1] = *(const v8h*)(arow + k0 + 8 * hi + 16);
            // B 32x16 f16 layout: halves 0-7 -> K = k0+8*hi+0..7, halves 8-15 -> K = k0+8*hi+16..23
            b.h[0] = *(const v8h*)(wrow + k0 + 8 * hi);
            b.h[1] = *(const v8h*)(wrow + k0 + 8 * hi + 16);
            acc = __builtin_amdgcn_wmma_f32_16x16x32_f16(
                false, a.v, false, b.v, (short)0, acc, false, false);
        }
        const float bv = bias ? bias[n] : 0.0f;
#pragma unroll
        for (int i = 0; i < 8; ++i) {           // C/D: M = i + 8*hi
            sOut[(i + 8 * hi) * N + n] = acc[i] + bv;
        }
    }
}

// Keras GRU (reset_after=True, both activations sigmoid) gate fusion.
// sXP/sHP: 16 x 3U (z | r | h). sH: fp32 state, sHh: f16 mirror for WMMA A.
template <int U>
__device__ __forceinline__ void gru_gates(const float* __restrict__ sXP,
                                          const float* __restrict__ sHP,
                                          float* __restrict__ sH,
                                          _Float16* __restrict__ sHh,
                                          int tid) {
    constexpr int N = 3 * U;
    for (int idx = tid; idx < 16 * U; idx += 256) {
        const int m = idx / U, u = idx - m * U;
        const float xz = sXP[m * N + u];
        const float xr = sXP[m * N + U + u];
        const float xh = sXP[m * N + 2 * U + u];
        const float hz = sHP[m * N + u];
        const float hr = sHP[m * N + U + u];
        const float hh = sHP[m * N + 2 * U + u];
        const float z  = 1.0f / (1.0f + __expf(-(xz + hz)));
        const float r  = 1.0f / (1.0f + __expf(-(xr + hr)));
        const float hc = 1.0f / (1.0f + __expf(-(xh + r * hh)));
        const float hn = z * sH[idx] + (1.0f - z) * hc;
        sH[idx]  = hn;
        sHh[idx] = (_Float16)hn;
    }
}

// Fully fused 3-layer GRU: one block = 16 batch rows, loop over all 32 steps.
__global__ __launch_bounds__(256) void gru3_fused(
    const float* __restrict__ noise,
    const float* __restrict__ b1, const float* __restrict__ b2,
    const float* __restrict__ b3,
    const _Float16* __restrict__ k1t, const _Float16* __restrict__ r1t,
    const _Float16* __restrict__ k2t, const _Float16* __restrict__ r2t,
    const _Float16* __restrict__ k3t, const _Float16* __restrict__ r3t,
    float* __restrict__ out) {
    __shared__ alignas(16) _Float16 sXT[16 * 128];   // x_t tile (f16)
    __shared__ alignas(16) _Float16 sH1h[16 * 256];  // f16 mirrors of states
    __shared__ alignas(16) _Float16 sH2h[16 * 64];
    __shared__ alignas(16) _Float16 sH3h[16 * 32];
    __shared__ float sH1[16 * 256];                  // fp32 carried states
    __shared__ float sH2[16 * 64];
    __shared__ float sH3[16 * 32];
    __shared__ alignas(16) float sXP[16 * 768];      // x-projection scratch
    __shared__ alignas(16) float sHP[16 * 768];      // h-projection scratch

    const int tid = threadIdx.x;
    const int r0  = blockIdx.x * 16;

    for (int i = tid; i < 16 * 256; i += 256) { sH1[i] = 0.f; sH1h[i] = (_Float16)0.f; }
    for (int i = tid; i < 16 * 64;  i += 256) { sH2[i] = 0.f; sH2h[i] = (_Float16)0.f; }
    for (int i = tid; i < 16 * 32;  i += 256) { sH3[i] = 0.f; sH3h[i] = (_Float16)0.f; }
    __syncthreads();

    for (int t = 0; t < 32; ++t) {
        // Stage x_t (16 x 128) -> LDS f16, coalesced along D.
        for (int i = tid; i < 16 * 128; i += 256) {
            const int m = i >> 7, d = i & 127;
            sXT[i] = (_Float16)noise[((size_t)(r0 + m) * 32 + t) * 128 + d];
        }
        __syncthreads();

        // ---- Layer 1 (128 -> 256) ----
        gemm16<128, 768>(sXT,  k1t, sXP, b1,        tid);   // x_t @ K1 + b_in
        gemm16<256, 768>(sH1h, r1t, sHP, b1 + 768,  tid);   // h1  @ R1 + b_rec
        __syncthreads();
        gru_gates<256>(sXP, sHP, sH1, sH1h, tid);
        __syncthreads();

        // ---- Layer 2 (256 -> 64) ----
        gemm16<256, 192>(sH1h, k2t, sXP, b2,        tid);
        gemm16<64,  192>(sH2h, r2t, sHP, b2 + 192,  tid);
        __syncthreads();
        gru_gates<64>(sXP, sHP, sH2, sH2h, tid);
        __syncthreads();

        // ---- Layer 3 (64 -> 32) ----
        gemm16<64, 96>(sH2h, k3t, sXP, b3,          tid);
        gemm16<32, 96>(sH3h, r3t, sHP, b3 + 96,     tid);
        __syncthreads();
        gru_gates<32>(sXP, sHP, sH3, sH3h, tid);
        __syncthreads();
    }

    // Final state of layer 3 -> output (B, 32), fp32.
    for (int i = tid; i < 16 * 32; i += 256) {
        out[(size_t)(r0 + (i >> 5)) * 32 + (i & 31)] = sH3[i];
    }
}

// Pre-pass: fp32 (K x N) weight -> f16 transposed Wt[n*K + k].
__global__ void convert_wt(const float* __restrict__ in,
                           _Float16* __restrict__ outw, int K, int N) {
    const int idx = blockIdx.x * blockDim.x + threadIdx.x;
    if (idx < K * N) {
        const int k = idx / N, n = idx - k * N;
        outw[(size_t)n * K + k] = (_Float16)in[idx];
    }
}

extern "C" void kernel_launch(void* const* d_in, const int* in_sizes, int n_in,
                              void* d_out, int out_size, void* d_ws, size_t ws_size,
                              hipStream_t stream) {
    const float* noise = (const float*)d_in[0];
    const float* k1 = (const float*)d_in[1];
    const float* r1 = (const float*)d_in[2];
    const float* b1 = (const float*)d_in[3];
    const float* k2 = (const float*)d_in[4];
    const float* r2 = (const float*)d_in[5];
    const float* b2 = (const float*)d_in[6];
    const float* k3 = (const float*)d_in[7];
    const float* r3 = (const float*)d_in[8];
    const float* b3 = (const float*)d_in[9];

    _Float16* ws  = (_Float16*)d_ws;
    _Float16* k1t = ws;                    // 768*128
    _Float16* r1t = k1t + 768 * 128;       // 768*256
    _Float16* k2t = r1t + 768 * 256;       // 192*256
    _Float16* r2t = k2t + 192 * 256;       // 192*64
    _Float16* k3t = r2t + 192 * 64;        // 96*64
    _Float16* r3t = k3t + 96 * 64;         // 96*32

    auto conv = [&](const float* src, int K, int N, _Float16* dst) {
        const int n = K * N;
        convert_wt<<<(n + 255) / 256, 256, 0, stream>>>(src, dst, K, N);
    };
    conv(k1, 128, 768, k1t);
    conv(r1, 256, 768, r1t);
    conv(k2, 256, 192, k2t);
    conv(r2, 64, 192, r2t);
    conv(k3, 64, 96, k3t);
    conv(r3, 32, 96, r3t);

    gru3_fused<<<4096 / 16, 256, 0, stream>>>(noise, b1, b2, b3,
                                              k1t, r1t, k2t, r2t, k3t, r3t,
                                              (float*)d_out);
}
